// PVConv_a_58428735094790
// MI455X (gfx1250) — matrix operand order repaired and checked by reading, not compiled, not run on hardware
//
#include <hip/hip_runtime.h>
#include <hip/hip_bf16.h>

typedef __attribute__((ext_vector_type(16))) _Float16 v16h;
typedef __attribute__((ext_vector_type(8)))  _Float16 v8h;
typedef __attribute__((ext_vector_type(8)))  float    v8f;

#define RDIM 32
#define RR   1024
#define R3   32768
#define NPT  65536
#define NB   4
// 1/sqrt(1 + 1e-5)
#define BN_RSQ 0.9999950000374997f

// ---------------- workspace layout (float offsets) ----------------
// all multiples of 16 floats -> 64B aligned
static constexpr size_t OFF_G0   = 0;                       // 4*64*32768
static constexpr size_t OFF_G1   = OFF_G0 + 8388608ull;
static constexpr size_t OFF_VF   = OFF_G1 + 8388608ull;     // vfeat 4*64*65536
static constexpr size_t OFF_PF   = OFF_VF + 16777216ull;    // pfeat
static constexpr size_t OFF_CN   = OFF_PF + 16777216ull;    // cnorm 4*3*65536
static constexpr size_t OFF_CT   = OFF_CN + 786432ull;      // counts 4*32768
static constexpr size_t OFF_MEAN = OFF_CT + 131072ull;      // 16
static constexpr size_t OFF_MAXN = OFF_MEAN + 16ull;        // 16
static constexpr size_t OFF_VA   = OFF_MAXN + 16ull;        // 64
static constexpr size_t OFF_VB   = OFF_VA + 64ull;          // 64
static constexpr size_t OFF_QC   = OFF_VB + 64ull;          // 16
static constexpr size_t OFF_WP   = OFF_QC + 16ull;          // f16: 5*27*2*64*32 = 552960 halves = 276480 floats
static constexpr size_t OFF_PP   = OFF_WP + 276480ull;      // f16: 2*64*32 = 4096 halves = 2048 floats
// total ~51.5M floats (~206 MB)

// ---------------- utility kernels ----------------
__global__ void k_zero(float* __restrict__ p, int n) {
  int i = blockIdx.x * 256 + threadIdx.x;
  if (i < n) p[i] = 0.0f;
}

__global__ __launch_bounds__(256) void k_mean(const float* __restrict__ coords,
                                              float* __restrict__ mean) {
  __shared__ float sm[256];
  int bd = blockIdx.x;                       // b*3 + dim, 12 blocks
  float s = 0.0f;
  for (int p = threadIdx.x; p < NPT; p += 256)
    s += coords[(size_t)bd * NPT + p];
  sm[threadIdx.x] = s; __syncthreads();
  for (int st = 128; st > 0; st >>= 1) {
    if (threadIdx.x < st) sm[threadIdx.x] += sm[threadIdx.x + st];
    __syncthreads();
  }
  if (threadIdx.x == 0) mean[bd] = sm[0] * (1.0f / NPT);
}

__global__ __launch_bounds__(256) void k_maxnorm(const float* __restrict__ coords,
                                                 const float* __restrict__ mean,
                                                 float* __restrict__ maxn) {
  __shared__ float sm[256];
  int b = blockIdx.x;                        // 4 blocks
  float m0 = mean[b*3+0], m1 = mean[b*3+1], m2 = mean[b*3+2];
  float mx = 0.0f;
  const float* cb = coords + (size_t)b * 3 * NPT;
  for (int p = threadIdx.x; p < NPT; p += 256) {
    float x = cb[p] - m0;
    float y = cb[NPT + p] - m1;
    float z = cb[2*NPT + p] - m2;
    mx = fmaxf(mx, x*x + y*y + z*z);
  }
  sm[threadIdx.x] = mx; __syncthreads();
  for (int st = 128; st > 0; st >>= 1) {
    if (threadIdx.x < st) sm[threadIdx.x] = fmaxf(sm[threadIdx.x], sm[threadIdx.x + st]);
    __syncthreads();
  }
  if (threadIdx.x == 0) maxn[b] = sqrtf(sm[0]);
}

__global__ __launch_bounds__(256) void k_voxelize(const float* __restrict__ coords,
                                                  const float* __restrict__ feats,
                                                  const float* __restrict__ mean,
                                                  const float* __restrict__ maxn,
                                                  float* __restrict__ cnorm,
                                                  float* __restrict__ sums,
                                                  float* __restrict__ counts) {
  int i = blockIdx.x * 256 + threadIdx.x;    // b*NPT + pt
  if (i >= NB * NPT) return;
  int b = i >> 16, pt = i & (NPT - 1);
  float inv = 1.0f / (maxn[b] * 2.0f);
  int vidx = 0;
  #pragma unroll
  for (int d = 0; d < 3; ++d) {
    float c = coords[((size_t)b*3 + d) * NPT + pt] - mean[b*3 + d];
    c = c * inv + 0.5f;
    c = c * 32.0f;
    c = fminf(fmaxf(c, 0.0f), 31.0f);
    cnorm[((size_t)b*3 + d) * NPT + pt] = c;
    int v = (int)rintf(c);                   // round-to-nearest-even, like jnp.round
    vidx = vidx * RDIM + v;                  // vx*1024 + vy*32 + vz
  }
  atomicAdd(&counts[(size_t)b * R3 + vidx], 1.0f);
  size_t fb = (size_t)b * 64 * NPT + pt;
  size_t sb = (size_t)b * 64 * R3 + vidx;
  for (int c = 0; c < 64; ++c)
    atomicAdd(&sums[sb + (size_t)c * R3], feats[fb + (size_t)c * NPT]);
}

__global__ __launch_bounds__(256) void k_divide(float* __restrict__ g,
                                                const float* __restrict__ counts) {
  int i = blockIdx.x * 256 + threadIdx.x;    // 4*64*32768
  if (i >= NB * 64 * R3) return;
  int b = i / (64 * R3);
  int v = i & (R3 - 1);
  float c = fmaxf(counts[(size_t)b * R3 + v], 1.0f);
  g[i] = g[i] / c;
}

// convert conv weights fp32 [O,I,3,3,3] -> f16 WMMA-A staging [layer][tap][cc][o][k32]
__global__ void k_wprep(const float* __restrict__ vw, const float* __restrict__ rw,
                        _Float16* __restrict__ wp) {
  int e = blockIdx.x * 256 + threadIdx.x;
  if (e >= 5 * 27 * 2 * 64 * 32) return;
  int t = e;
  int k  = t & 31;  t >>= 5;
  int o  = t & 63;  t >>= 6;
  int cc = t & 1;   t >>= 1;
  int tap = t % 27; int l = t / 27;
  const float* src = (l == 0) ? vw : rw + (size_t)(l - 1) * 64 * 64 * 27;
  wp[e] = (_Float16)src[((size_t)o * 64 + cc * 32 + k) * 27 + tap];
}

__global__ void k_pfprep(const float* __restrict__ pw, _Float16* __restrict__ pp) {
  int e = blockIdx.x * 256 + threadIdx.x;
  if (e >= 2 * 64 * 32) return;
  int k = e & 31;
  int o = (e >> 5) & 63;
  int cc = e >> 11;
  pp[e] = (_Float16)pw[(size_t)o * 64 + cc * 32 + k];
}

// fold attention head: q = va.vfeat + vb.pfeat + qconst (exact algebra of the reference)
__global__ void k_attnprep(const float* __restrict__ wa, const float* __restrict__ ba,
                           const float* __restrict__ wb, const float* __restrict__ bb,
                           const float* __restrict__ wc, const float* __restrict__ bc,
                           float* __restrict__ va, float* __restrict__ vb,
                           float* __restrict__ qc) {
  int t = threadIdx.x;
  if (t < 64) {
    float sa = 0.0f, sb = 0.0f;
    for (int j = 0; j < 16; ++j) {
      sa += wc[j]      * wa[j * 64 + t];
      sb += wc[16 + j] * wb[j * 64 + t];
    }
    va[t] = sa; vb[t] = sb;
  }
  if (t == 0) {
    float s = bc[0];
    for (int j = 0; j < 16; ++j) s += wc[j] * ba[j] + wc[16 + j] * bb[j];
    qc[0] = s;
  }
}

// ---------------- WMMA 3x3x3 conv + bias + BN (+residual) + ReLU ----------------
// block = 256 thr = 8 waves; one (b,d0,d1) row of 32 voxels, all 64 out channels.
// wave wv: m-tile = wv&3 (16 out ch), x-tile = wv>>2 (16 voxels).
__global__ __launch_bounds__(256) void k_conv3d(const float* __restrict__ gin,
                                                const _Float16* __restrict__ wl,
                                                const float* __restrict__ cb,
                                                const float* __restrict__ bng,
                                                const float* __restrict__ bnb,
                                                const float* __restrict__ resid,
                                                float* __restrict__ gout,
                                                int do_res) {
  __shared__ __align__(32) _Float16 s_in[9 * 34 * 64];  // [a0*3+a1][x+1][c] f16
  const int b  = blockIdx.x >> 10;
  const int d0 = (blockIdx.x >> 5) & 31;
  const int d1 = blockIdx.x & 31;
  const int tid = threadIdx.x;
  const size_t base_in = (size_t)b * 64 * R3;

  // stage 3x3 neighborhood slab, x contiguous for coalescing, store [slab][x][c]
  for (int e = tid; e < 9 * 64 * 34; e += 256) {
    int xi  = e % 34;
    int t   = e / 34;
    int c   = t & 63;
    int a03 = t >> 6;                 // 0..8  (a0*3 + a1)
    int a1 = a03 % 3, a0 = a03 / 3;
    int g0i = d0 + a0 - 1, g1i = d1 + a1 - 1, g2i = xi - 1;
    float v = 0.0f;
    if ((unsigned)g0i < 32u && (unsigned)g1i < 32u && (unsigned)g2i < 32u)
      v = gin[base_in + (size_t)c * R3 + g0i * RR + g1i * RDIM + g2i];
    s_in[(a03 * 34 + xi) * 64 + c] = (_Float16)v;
  }
  __syncthreads();

  const int lane = tid & 31, wv = tid >> 5;
  const int mt = wv & 3, xt = wv >> 2;
  const int hf = lane >> 4, nn = lane & 15;

  v8f acc = {};
  for (int tap = 0; tap < 27; ++tap) {
    int k0 = tap / 9, rem = tap % 9;
    int k1 = rem / 3, k2 = rem % 3;
    int ldsrow = (k0 * 3 + k1) * 34;
    #pragma unroll
    for (int cc = 0; cc < 2; ++cc) {
      // A fragment: lanes 0-15 k = 0..15 ; lanes 16-31 k = 16..31
      const _Float16* wp = wl + ((size_t)(tap * 2 + cc) * 64 + (mt * 16 + nn)) * 32;
      v8h a0v = *(const v8h*)(wp + 16 * hf);
      v8h a1v = *(const v8h*)(wp + 16 * hf + 8);
      v16h a = __builtin_shufflevector(a0v, a1v, 0,1,2,3,4,5,6,7,8,9,10,11,12,13,14,15);
      // B fragment: lane group picks k = 16*hf..16*hf+15, contiguous channels in LDS
      int xi = xt * 16 + nn + k2;     // (output x) + k2 - 1, plus the +1 halo offset
      const v16h bf = *(const v16h*)(&s_in[(ldsrow + xi) * 64 + cc * 32 + 16 * hf]);
      acc = __builtin_amdgcn_wmma_f32_16x16x32_f16(false, a, false, bf,
                                                   (short)0, acc, false, false);
    }
  }

  // epilogue: D layout m = r + 8*hf, n = nn
  const size_t obase = (size_t)b * 64 * R3 + (size_t)d0 * RR + d1 * RDIM + (xt * 16 + nn);
  #pragma unroll
  for (int r = 0; r < 8; ++r) {
    int o = mt * 16 + r + 8 * hf;
    float y = (acc[r] + cb[o]) * (bng[o] * BN_RSQ) + bnb[o];
    size_t oi = obase + (size_t)o * R3;
    if (do_res) y += resid[oi];      // in-place safe: same element, same thread
    gout[oi] = fmaxf(y, 0.0f);
  }
}

// ---------------- trilinear devoxelize ----------------
__global__ __launch_bounds__(256) void k_devox(const float* __restrict__ g,
                                               const float* __restrict__ cn,
                                               float* __restrict__ vf) {
  int i = blockIdx.x * 256 + threadIdx.x;
  if (i >= NB * NPT) return;
  int b = i >> 16, pt = i & (NPT - 1);
  float c0 = cn[((size_t)b*3 + 0) * NPT + pt];
  float c1 = cn[((size_t)b*3 + 1) * NPT + pt];
  float c2 = cn[((size_t)b*3 + 2) * NPT + pt];
  int l0 = min((int)floorf(c0), 31), l1 = min((int)floorf(c1), 31), l2 = min((int)floorf(c2), 31);
  int h0 = min(l0 + 1, 31), h1 = min(l1 + 1, 31), h2 = min(l2 + 1, 31);
  float f0 = c0 - (float)l0, f1 = c1 - (float)l1, f2 = c2 - (float)l2;
  int   idx8[8];
  float w8[8];
  int k = 0;
  #pragma unroll
  for (int ax = 0; ax < 2; ++ax)
    #pragma unroll
    for (int ay = 0; ay < 2; ++ay)
      #pragma unroll
      for (int az = 0; az < 2; ++az) {
        int ix = ax ? h0 : l0, iy = ay ? h1 : l1, iz = az ? h2 : l2;
        float wx = ax ? f0 : 1.0f - f0;
        float wy = ay ? f1 : 1.0f - f1;
        float wz = az ? f2 : 1.0f - f2;
        idx8[k] = ix * RR + iy * RDIM + iz;
        w8[k] = wx * wy * wz;
        ++k;
      }
  size_t gb = (size_t)b * 64 * R3;
  size_t ob = (size_t)b * 64 * NPT + pt;
  for (int c = 0; c < 64; ++c) {
    const float* gc = g + gb + (size_t)c * R3;
    float acc = 0.0f;
    #pragma unroll
    for (int j = 0; j < 8; ++j) acc += w8[j] * gc[idx8[j]];
    vf[ob + (size_t)c * NPT] = acc;
  }
}

// ---------------- point MLP (64x64) via WMMA: pfeat = relu(bn(W f + b)) ----------------
// block = 256 thr = 8 waves; tile: 64 out ch x 32 points per workgroup.
__global__ __launch_bounds__(256) void k_pfeat(const float* __restrict__ feats,
                                               const _Float16* __restrict__ pp,
                                               const float* __restrict__ pb,
                                               const float* __restrict__ bng,
                                               const float* __restrict__ bnb,
                                               float* __restrict__ pf) {
  __shared__ __align__(32) _Float16 s_f[32 * 64];   // [pt][c] f16
  const int b   = blockIdx.x >> 11;                 // 2048 blocks per batch
  const int pt0 = (blockIdx.x & 2047) * 32;
  const int tid = threadIdx.x;

  for (int e = tid; e < 2048; e += 256) {
    int pt = e & 31, c = e >> 5;
    float v = feats[((size_t)b * 64 + c) * NPT + pt0 + pt];
    s_f[pt * 64 + c] = (_Float16)v;
  }
  __syncthreads();

  const int lane = tid & 31, wv = tid >> 5;
  const int mt = wv & 3, ptile = wv >> 2;
  const int hf = lane >> 4, nn = lane & 15;
  const int ptl = ptile * 16 + nn;

  v8f acc = {};
  #pragma unroll
  for (int cc = 0; cc < 2; ++cc) {
    const _Float16* wp = pp + ((size_t)cc * 64 + (mt * 16 + nn)) * 32;
    v8h a0v = *(const v8h*)(wp + 16 * hf);
    v8h a1v = *(const v8h*)(wp + 16 * hf + 8);
    v16h a = __builtin_shufflevector(a0v, a1v, 0,1,2,3,4,5,6,7,8,9,10,11,12,13,14,15);
    const v16h bf = *(const v16h*)(&s_f[ptl * 64 + cc * 32 + 16 * hf]);
    acc = __builtin_amdgcn_wmma_f32_16x16x32_f16(false, a, false, bf,
                                                 (short)0, acc, false, false);
  }
  #pragma unroll
  for (int r = 0; r < 8; ++r) {
    int o = mt * 16 + r + 8 * hf;
    float y = (acc[r] + pb[o]) * (bng[o] * BN_RSQ) + bnb[o];
    pf[((size_t)b * 64 + o) * NPT + pt0 + ptl] = fmaxf(y, 0.0f);
  }
}

// ---------------- fused attention + output ----------------
__global__ __launch_bounds__(256) void k_fuse(const float* __restrict__ vf,
                                              const float* __restrict__ pf,
                                              const float* __restrict__ va,
                                              const float* __restrict__ vb,
                                              const float* __restrict__ qc,
                                              float* __restrict__ out) {
  int i = blockIdx.x * 256 + threadIdx.x;
  if (i >= NB * NPT) return;
  int b = i >> 16, pt = i & (NPT - 1);
  size_t base = (size_t)b * 64 * NPT + pt;
  float q = qc[0];
  for (int c = 0; c < 64; ++c) {
    size_t idx = base + (size_t)c * NPT;
    q += va[c] * vf[idx] + vb[c] * pf[idx];
  }
  float att = 1.0f + 1.0f / (1.0f + expf(-q));
  for (int c = 0; c < 64; ++c) {
    size_t idx = base + (size_t)c * NPT;
    out[idx] = vf[idx] * att + pf[idx];
  }
}

// ---------------- launcher ----------------
extern "C" void kernel_launch(void* const* d_in, const int* in_sizes, int n_in,
                              void* d_out, int out_size, void* d_ws, size_t ws_size,
                              hipStream_t stream) {
  const float* features = (const float*)d_in[0];
  const float* coords   = (const float*)d_in[1];
  const float* vconv_w  = (const float*)d_in[2];
  const float* vconv_b  = (const float*)d_in[3];
  const float* vbn_g    = (const float*)d_in[4];
  const float* vbn_b    = (const float*)d_in[5];
  const float* res_w    = (const float*)d_in[6];
  const float* res_b    = (const float*)d_in[7];
  const float* res_bn_g = (const float*)d_in[8];
  const float* res_bn_b = (const float*)d_in[9];
  const float* pf_w     = (const float*)d_in[10];
  const float* pf_b     = (const float*)d_in[11];
  const float* pf_bn_g  = (const float*)d_in[12];
  const float* pf_bn_b  = (const float*)d_in[13];
  const float* attn_wa  = (const float*)d_in[14];
  const float* attn_ba  = (const float*)d_in[15];
  const float* attn_wb  = (const float*)d_in[16];
  const float* attn_bb  = (const float*)d_in[17];
  const float* attn_wc  = (const float*)d_in[18];
  const float* attn_bc  = (const float*)d_in[19];

  float* ws = (float*)d_ws;
  float* g0     = ws + OFF_G0;
  float* g1     = ws + OFF_G1;
  float* vfeat  = ws + OFF_VF;
  float* pfeat  = ws + OFF_PF;
  float* cnorm  = ws + OFF_CN;
  float* counts = ws + OFF_CT;
  float* meanp  = ws + OFF_MEAN;
  float* maxnp  = ws + OFF_MAXN;
  float* va     = ws + OFF_VA;
  float* vb     = ws + OFF_VB;
  float* qc     = ws + OFF_QC;
  _Float16* wprep  = (_Float16*)(ws + OFF_WP);
  _Float16* pfprep = (_Float16*)(ws + OFF_PP);

  // zero scatter targets (harness does not re-zero between replays)
  k_zero<<<(8388608 + 255) / 256, 256, 0, stream>>>(g0, 8388608);
  k_zero<<<(131072 + 255) / 256, 256, 0, stream>>>(counts, 131072);

  // voxelization preliminaries
  k_mean<<<12, 256, 0, stream>>>(coords, meanp);
  k_maxnorm<<<4, 256, 0, stream>>>(coords, meanp, maxnp);
  k_voxelize<<<(NB * NPT) / 256, 256, 0, stream>>>(coords, features, meanp, maxnp,
                                                   cnorm, g0, counts);
  k_divide<<<(NB * 64 * R3) / 256, 256, 0, stream>>>(g0, counts);

  // weight staging (f16 WMMA layouts) + attention folding
  k_wprep<<<(5 * 27 * 2 * 64 * 32 + 255) / 256, 256, 0, stream>>>(vconv_w, res_w, wprep);
  k_pfprep<<<16, 256, 0, stream>>>(pf_w, pfprep);
  k_attnprep<<<1, 64, 0, stream>>>(attn_wa, attn_ba, attn_wb, attn_bb, attn_wc, attn_bc,
                                   va, vb, qc);

  const int WL = 27 * 2 * 64 * 32;  // per-layer f16 weight stride
  // BasicConvolutionBlock
  k_conv3d<<<4096, 256, 0, stream>>>(g0, wprep + 0 * WL, vconv_b, vbn_g, vbn_b,
                                     nullptr, g1, 0);
  // ResidualBlock 0
  k_conv3d<<<4096, 256, 0, stream>>>(g1, wprep + 1 * WL, res_b + 0, res_bn_g + 0,
                                     res_bn_b + 0, nullptr, g0, 0);
  k_conv3d<<<4096, 256, 0, stream>>>(g0, wprep + 2 * WL, res_b + 64, res_bn_g + 64,
                                     res_bn_b + 64, g1, g1, 1);
  // ResidualBlock 1
  k_conv3d<<<4096, 256, 0, stream>>>(g1, wprep + 3 * WL, res_b + 128, res_bn_g + 128,
                                     res_bn_b + 128, nullptr, g0, 0);
  k_conv3d<<<4096, 256, 0, stream>>>(g0, wprep + 4 * WL, res_b + 192, res_bn_g + 192,
                                     res_bn_b + 192, g1, g1, 1);

  // point branch + fusion
  k_devox<<<(NB * NPT) / 256, 256, 0, stream>>>(g1, cnorm, vfeat);
  k_pfeat<<<NB * (NPT / 32), 256, 0, stream>>>(features, pfprep, pf_b, pf_bn_g,
                                               pf_bn_b, pfeat);
  k_fuse<<<(NB * NPT) / 256, 256, 0, stream>>>(vfeat, pfeat, va, vb, qc, (float*)d_out);
}
